// GPS_prompt_mt_2roi_76974403879198
// MI455X (gfx1250) — matrix operand order repaired and checked by reading, not compiled, not run on hardware
//
#include <hip/hip_runtime.h>
#include <cstddef>

// ---------------- problem constants (from the reference) ----------------
#define NNODES 8448      // B * ROI
#define NB     32
#define LROI   264
#define HCC    264
#define HC2C   256
#define EPER   8448
#define SEGW   8576      // E_PER + SUB
#define SUBW   128
#define NEDGE  270336    // NB * EPER
#define NSEG   278784    // NEDGE + NNODES (self loops)
#define KSEL   80
#define NHEAD  4
#define XW     456       // ROI + 3*RPT

// d_out layout (return-tuple order, flattened float)
#define O_X1O   0
#define O_SP1   32
#define O_X2O   64
#define O_SP2   128
#define O_X1L   192
#define O_SUB1  8384
#define O_X2L   16576
#define O_SUB2  24768
#define O_PERM1 32960
#define O_SC1   35520
#define O_PERM2 38080
#define O_SC2   40640
#define O_ATT   43200

typedef __attribute__((ext_vector_type(16))) __bf16 v16bf;
typedef __attribute__((ext_vector_type(8)))  float  v8f;

__device__ __forceinline__ __bf16 f2bf(float f) {
  union { float f; unsigned u; } a; a.f = f;
  unsigned r = (a.u + 0x7FFFu + ((a.u >> 16) & 1u)) >> 16;   // RNE
  union { unsigned short s; __bf16 b; } o; o.s = (unsigned short)r;
  return o.b;
}
__device__ __forceinline__ float sigmoidf_(float x) { return 1.f / (1.f + __expf(-x)); }

__device__ __forceinline__ float atomicMaxF(float* addr, float val) {
  unsigned* ua = (unsigned*)addr;
  unsigned old = *ua;
  while (true) {
    float f = __uint_as_float(old);
    if (f >= val) break;
    unsigned assumed = old;
    old = atomicCAS(ua, assumed, __float_as_uint(val));
    if (old == assumed) break;
  }
  return __uint_as_float(old);
}

// ------- fp32 -> padded bf16 pack (batched; z-stride in elements) -------
__global__ void k_pack_bf16(const float* __restrict__ src, int lds_, int rows, int cols,
                            size_t szstride, __bf16* __restrict__ dst,
                            int rowsP, int colsP, int nz)
{
  int i = blockIdx.x * blockDim.x + threadIdx.x;
  int per = rowsP * colsP;
  if (i >= nz * per) return;
  int z = i / per, rem = i % per;
  int r = rem / colsP, c = rem % colsP;
  float v = 0.f;
  if (r < rows && c < cols) {
    const float* s = src + (size_t)z * szstride;
    v = s[(size_t)r * lds_ + c];
    if (c + 256 < cols) __builtin_prefetch(&s[(size_t)r * lds_ + c + 256], 0, 3);
  }
  dst[i] = f2bf(v);
}

// gather one head slice per z = b*NHEAD+h into padded bf16 (trans: V^T)
__global__ void k_pack_head(const float* __restrict__ src, int lds_, int c0, int dh,
                            __bf16* __restrict__ dst, int rowsP, int colsP, int nz, int trans)
{
  int i = blockIdx.x * blockDim.x + threadIdx.x;
  int per = rowsP * colsP;
  if (i >= nz * per) return;
  int z = i / per, rem = i % per;
  int r = rem / colsP, c = rem % colsP;
  int b = z / NHEAD, h = z % NHEAD;
  float v = 0.f;
  if (!trans) {                 // dst[z][l=r][d=c]
    if (r < LROI && c < dh) v = src[(size_t)(b * LROI + r) * lds_ + c0 + h * dh + c];
  } else {                      // dst[z][d=r][m=c]  (V transposed)
    if (c < LROI && r < dh) v = src[(size_t)(b * LROI + c) * lds_ + c0 + h * dh + r];
  }
  dst[i] = f2bf(v);
}

// ======================================================================
//  Batched GEMM:  C_z[M,N] = act( scale * A_z[M,K] @ W_z[N,K]^T + bias + res )
//  A, W pre-packed bf16 (A rows padded to 128, W rows to 64, K to 32,
//  16B aligned) -> zero predication in the hot loop. Tiles staged to LDS
//  with global_load_async_to_lds_b128 (ASYNCcnt), double buffered.
//  Tile 128x64x32, 8 waves; each wave owns 32x32 -> 2 A-frags x 2 B-frags
//  -> 4x v_wmma_f32_16x16x32_bf16 per K-step.
//  C offset per z: (z/nh)*strideCb + (z%nh)*strideCh (lets per-head output
//  columns land directly in the fused activation row).
// ======================================================================
#define TM 128
#define TN 64
#define TK 32

__global__ __launch_bounds__(256) void k_gemm_bf16(
    const __bf16* __restrict__ A, int lda, size_t strideA,
    const __bf16* __restrict__ W, int ldw, size_t strideW,
    const float* __restrict__ bias,
    const float* __restrict__ res, int ldres,
    float* __restrict__ C, int ldc, size_t strideCb, size_t strideCh, int nh,
    int M, int N, int Kp, int act, float scale)
{
  __shared__ alignas(16) __bf16 As[2][TM][TK + 8];
  __shared__ alignas(16) __bf16 Bs[2][TN][TK + 8];

  const int z = blockIdx.z;
  A += (size_t)z * strideA;
  W += (size_t)z * strideW;
  C += (size_t)(z / nh) * strideCb + (size_t)(z % nh) * strideCh;

  const int tid  = threadIdx.x;
  const int lane = tid & 31;
  const int wid  = tid >> 5;
  const int wm   = wid & 3;    // 32-row band
  const int wn   = wid >> 2;   // 32-col band
  const int m0   = blockIdx.y * TM;
  const int n0   = blockIdx.x * TN;

  // staging: A tile 128x32 = 512 x 16B chunks (2/thread), B tile 64x32 (1/thread)
  const int cr = tid >> 2;            // row 0..63
  const int cq = tid & 3;             // 16B chunk in row

  const unsigned rowB   = (TK + 8) * 2;                 // LDS bytes per row (80)
  const unsigned stageA = (unsigned)sizeof(As[0]);
  const unsigned stageB = (unsigned)sizeof(Bs[0]);
  const unsigned ldsA   = (unsigned)(size_t)&As[0][0][0] + (unsigned)cr * rowB + (unsigned)cq * 16u;
  const unsigned ldsB   = (unsigned)(size_t)&Bs[0][0][0] + (unsigned)cr * rowB + (unsigned)cq * 16u;
  const unsigned aByte0 = (unsigned)(((size_t)(m0 + cr)      * (size_t)lda + (size_t)cq * 8u) * 2u);
  const unsigned aByte1 = (unsigned)(((size_t)(m0 + cr + 64) * (size_t)lda + (size_t)cq * 8u) * 2u);
  const unsigned bByte  = (unsigned)(((size_t)(n0 + cr)      * (size_t)ldw + (size_t)cq * 8u) * 2u);

  const int nk = Kp / TK;

  auto issue = [&](int kt, int buf) {
    unsigned la0 = ldsA + (unsigned)buf * stageA;
    unsigned la1 = la0 + 64u * rowB;
    unsigned lb  = ldsB + (unsigned)buf * stageB;
    unsigned ko  = (unsigned)(kt * TK * 2);
    unsigned oa0 = aByte0 + ko, oa1 = aByte1 + ko, ob = bByte + ko;
    asm volatile("global_load_async_to_lds_b128 %0, %1, %2"
                 :: "v"(la0), "v"(oa0), "s"(A) : "memory");
    asm volatile("global_load_async_to_lds_b128 %0, %1, %2"
                 :: "v"(la1), "v"(oa1), "s"(A) : "memory");
    asm volatile("global_load_async_to_lds_b128 %0, %1, %2"
                 :: "v"(lb), "v"(ob), "s"(W) : "memory");
  };

  v8f acc00 = {}, acc01 = {}, acc10 = {}, acc11 = {};

  issue(0, 0);
  for (int kt = 0; kt < nk; ++kt) {
    const int buf = kt & 1;
    if (kt + 1 < nk) {
      issue(kt + 1, buf ^ 1);
      asm volatile("s_wait_asynccnt 0x3" ::: "memory");   // this tile's 3 copies done
    } else {
      asm volatile("s_wait_asynccnt 0x0" ::: "memory");
    }
    __syncthreads();

    // fragments: contiguous k = 16*(lane/16) + i, same order for A and B
    v16bf a0, a1, b0, b1;
    const int am    = wm * 32 + (lane & 15);
    const int khalf = (lane >> 4) * 16;
#pragma unroll
    for (int i = 0; i < 16; ++i) {
      int k = khalf + i;
      a0[i] = As[buf][am][k];
      a1[i] = As[buf][am + 16][k];
    }
    const int bn = wn * 32 + (lane & 15);
    const int kb = (lane >> 4) * 16;
#pragma unroll
    for (int i = 0; i < 16; ++i) {
      b0[i] = Bs[buf][bn][kb + i];
      b1[i] = Bs[buf][bn + 16][kb + i];
    }
    acc00 = __builtin_amdgcn_wmma_f32_16x16x32_bf16(false, a0, false, b0, (short)0, acc00, false, false);
    acc01 = __builtin_amdgcn_wmma_f32_16x16x32_bf16(false, a0, false, b1, (short)0, acc01, false, false);
    acc10 = __builtin_amdgcn_wmma_f32_16x16x32_bf16(false, a1, false, b0, (short)0, acc10, false, false);
    acc11 = __builtin_amdgcn_wmma_f32_16x16x32_bf16(false, a1, false, b1, (short)0, acc11, false, false);
    __syncthreads();
  }

  // C/D layout: VGPR v -> row v (+8 for lanes 16-31), col = lane&15
  const int rb = m0 + wm * 32 + ((lane >> 4) << 3);
  const int cb = n0 + wn * 32 + (lane & 15);
  v8f accs[2][2] = { { acc00, acc01 }, { acc10, acc11 } };
#pragma unroll
  for (int mi = 0; mi < 2; ++mi) {
#pragma unroll
    for (int v = 0; v < 8; ++v) {
      int r = rb + mi * 16 + v;
      if (r >= M) continue;
#pragma unroll
      for (int ti = 0; ti < 2; ++ti) {
        int c = cb + ti * 16;
        if (c >= N) continue;
        float xv = accs[mi][ti][v] * scale;
        if (bias) xv += bias[c];
        if (res)  xv += res[(size_t)r * ldres + c];
        if (act == 1)      xv = xv > 0.f ? xv : 0.f;
        else if (act == 2) xv = sigmoidf_(xv);
        C[(size_t)r * ldc + c] = xv;
      }
    }
  }
}

// ---------------------- elementwise / BN kernels ----------------------
__global__ void k_bn_in(const float* __restrict__ X, int ldx,
                        const float* __restrict__ g, const float* __restrict__ b,
                        float* __restrict__ Y, int rows, int cols)
{
  int i = blockIdx.x * blockDim.x + threadIdx.x;
  if (i >= rows * cols) return;
  int r = i / cols, c = i % cols;
  const float inv = 0.99999500f;   // 1/sqrt(1+1e-5)
  Y[i] = g[c] * (X[(size_t)r * ldx + c] * inv) + b[c];
}

__global__ void k_bn_res(const float* __restrict__ A, const float* __restrict__ R,
                         const float* __restrict__ g, const float* __restrict__ b,
                         float* __restrict__ Y, int rows, int cols, int relu)
{
  int i = blockIdx.x * blockDim.x + threadIdx.x;
  if (i >= rows * cols) return;
  int c = i % cols;
  const float inv = 0.99999500f;
  float v = A[i] + (R ? R[i] : 0.f);
  v = g[c] * (v * inv) + b[c];
  if (relu) v = v > 0.f ? v : 0.f;
  Y[i] = v;
}

__global__ void k_add(const float* a, const float* b, float* y, int n)
{
  int i = blockIdx.x * blockDim.x + threadIdx.x;
  if (i < n) y[i] = a[i] + b[i];
}

__global__ void k_fill(float* p, float v, int n)
{
  int i = blockIdx.x * blockDim.x + threadIdx.x;
  if (i < n) p[i] = v;
}

__global__ void k_copy(const float* s, float* d, int n)
{
  int i = blockIdx.x * blockDim.x + threadIdx.x;
  if (i < n) d[i] = s[i];
}

__global__ void k_copy2d(const float* __restrict__ s, int lds_,
                         float* __restrict__ d, int ldd, int rows, int cols)
{
  int i = blockIdx.x * blockDim.x + threadIdx.x;
  if (i >= rows * cols) return;
  int r = i / cols, c = i % cols;
  d[(size_t)r * ldd + c] = s[(size_t)r * lds_ + c];
}

// ---------------------- GINE message + scatter ------------------------
__global__ __launch_bounds__(256) void k_gine_msg(
    const float* __restrict__ x0, const int* __restrict__ src, const int* __restrict__ dst,
    const float* __restrict__ ew, const float* __restrict__ lew, const float* __restrict__ leb,
    float* __restrict__ agg)
{
  int e = blockIdx.x;
  if (e >= NEDGE) return;
  int s = src[e], d = dst[e];
  float ea = ew[(size_t)(e / EPER) * SEGW + (e % EPER)];
  const float* xs = x0 + (size_t)s * HCC;
  float* ad = agg + (size_t)d * HCC;
  for (int c = threadIdx.x; c < HCC; c += blockDim.x) {
    float m = xs[c] + lew[c] * ea + leb[c];
    m = m > 0.f ? m : 0.f;
    atomicAdd(&ad[c], m);
  }
}

// ---------------------- softmax over a 264-wide row -------------------
__global__ __launch_bounds__(256) void k_softmax_row(float* __restrict__ P)
{
  float* pr = P + (size_t)blockIdx.x * LROI;
  int tid = threadIdx.x;
  __shared__ float red[256];
  __shared__ float smax, ssum;
  float v0 = (tid < LROI) ? pr[tid] : -3.4e38f;
  float v1 = (tid + 256 < LROI) ? pr[tid + 256] : -3.4e38f;
  red[tid] = fmaxf(v0, v1); __syncthreads();
  for (int st = 128; st; st >>= 1) { if (tid < st) red[tid] = fmaxf(red[tid], red[tid + st]); __syncthreads(); }
  if (tid == 0) smax = red[0];
  __syncthreads();
  float e0 = (tid < LROI) ? __expf(v0 - smax) : 0.f;
  float e1 = (tid + 256 < LROI) ? __expf(v1 - smax) : 0.f;
  red[tid] = e0 + e1; __syncthreads();
  for (int st = 128; st; st >>= 1) { if (tid < st) red[tid] += red[tid + st]; __syncthreads(); }
  if (tid == 0) ssum = red[0];
  __syncthreads();
  float inv = 1.f / ssum;
  if (tid < LROI) pr[tid] = e0 * inv;
  if (tid + 256 < LROI) pr[tid + 256] = e1 * inv;
}

__global__ void k_attn_mean(const float* __restrict__ P, float* __restrict__ out)
{
  int i = blockIdx.x * blockDim.x + threadIdx.x;
  const int LL = LROI * LROI;
  if (i >= NB * LL) return;
  int lm = i % LL, b = i / LL;
  float acc = 0.f;
  for (int h = 0; h < NHEAD; ++h) acc += P[(size_t)(b * NHEAD + h) * LL + lm];
  out[i] = acc * (1.f / NHEAD);
}

// ------------------------------- GAT ----------------------------------
__global__ void k_rowdot2(const float* __restrict__ xt,
                          const float* __restrict__ as_, const float* __restrict__ ad_,
                          float* __restrict__ es, float* __restrict__ ed, int rows, int cols)
{
  int r = blockIdx.x * blockDim.x + threadIdx.x;
  if (r >= rows) return;
  const float* row = xt + (size_t)r * cols;
  float a = 0.f, b = 0.f;
  for (int c = 0; c < cols; ++c) { a += row[c] * as_[c]; b += row[c] * ad_[c]; }
  es[r] = a; ed[r] = b;
}

__global__ void k_gat_edge1(const int* __restrict__ src, const int* __restrict__ dst,
                            const float* __restrict__ es, const float* __restrict__ ed,
                            float* __restrict__ e_out, float* __restrict__ nmax)
{
  int i = blockIdx.x * blockDim.x + threadIdx.x;
  if (i >= NSEG) return;
  int s = (i < NEDGE) ? src[i] : (i - NEDGE);
  int d = (i < NEDGE) ? dst[i] : (i - NEDGE);
  float e = es[s] + ed[d];
  e = e > 0.f ? e : 0.2f * e;
  e_out[i] = e;
  atomicMaxF(&nmax[d], e);
}

__global__ void k_gat_edge2(const int* __restrict__ dst, const float* __restrict__ e_in,
                            const float* __restrict__ nmax,
                            float* __restrict__ ex, float* __restrict__ nsum)
{
  int i = blockIdx.x * blockDim.x + threadIdx.x;
  if (i >= NSEG) return;
  int d = (i < NEDGE) ? dst[i] : (i - NEDGE);
  float v = __expf(e_in[i] - nmax[d]);
  ex[i] = v;
  atomicAdd(&nsum[d], v);
}

__global__ __launch_bounds__(256) void k_gat_scatter(
    const int* __restrict__ src, const int* __restrict__ dst,
    const float* __restrict__ ex, const float* __restrict__ nsum,
    const float* __restrict__ xt, float* __restrict__ gout)
{
  int i = blockIdx.x;
  if (i >= NSEG) return;
  int s = (i < NEDGE) ? src[i] : (i - NEDGE);
  int d = (i < NEDGE) ? dst[i] : (i - NEDGE);
  float alpha = ex[i] / nsum[d];
  const float* xs = xt + (size_t)s * HC2C;
  float* gd = gout + (size_t)d * HC2C;
  for (int c = threadIdx.x; c < HC2C; c += blockDim.x)
    atomicAdd(&gd[c], alpha * xs[c]);
}

__global__ void k_gat_fin(const float* __restrict__ gout, const float* __restrict__ bias,
                          float* __restrict__ y, int rows, int cols)
{
  int i = blockIdx.x * blockDim.x + threadIdx.x;
  if (i >= rows * cols) return;
  float v = gout[i] + bias[i % cols];
  y[i] = v > 0.f ? v : 0.f;
}

// --------------------------- pooling / top-k --------------------------
__global__ void k_vecnorm(const float* __restrict__ w, float* __restrict__ out, int n)
{
  __shared__ float red[256];
  float a = 0.f;
  for (int i = threadIdx.x; i < n; i += 256) a += w[i] * w[i];
  red[threadIdx.x] = a; __syncthreads();
  for (int st = 128; st; st >>= 1) { if (threadIdx.x < st) red[threadIdx.x] += red[threadIdx.x + st]; __syncthreads(); }
  if (threadIdx.x == 0) out[0] = sqrtf(red[0]);
}

__global__ void k_pool_score(const float* __restrict__ T, const float* __restrict__ w,
                             const float* __restrict__ normp, float* __restrict__ score,
                             int rows, int cols)
{
  int r = blockIdx.x * blockDim.x + threadIdx.x;
  if (r >= rows) return;
  float inv = 1.f / normp[0];
  const float* row = T + (size_t)r * cols;
  float acc = 0.f;
  for (int c = 0; c < cols; ++c) acc += row[c] * w[c];
  score[r] = sigmoidf_(acc * inv);
}

__global__ __launch_bounds__(256) void k_topk(
    const float* __restrict__ score, float* __restrict__ vals, int* __restrict__ perm,
    float* __restrict__ out_perm_f, float* __restrict__ out_sig)
{
  __shared__ float sv[LROI];
  __shared__ float rv[256];
  __shared__ int   ri[256];
  int b = blockIdx.x;
  int tid = threadIdx.x;
  for (int i = tid; i < LROI; i += 256) sv[i] = score[b * LROI + i];
  __syncthreads();
  for (int j = 0; j < KSEL; ++j) {
    float best = -3.4e38f; int bi = 0;
    for (int i = tid; i < LROI; i += 256) {
      float v = sv[i];
      if (v > best) { best = v; bi = i; }
    }
    rv[tid] = best; ri[tid] = bi;
    __syncthreads();
    for (int st = 128; st; st >>= 1) {
      if (tid < st) {
        if (rv[tid + st] > rv[tid] || (rv[tid + st] == rv[tid] && ri[tid + st] < ri[tid])) {
          rv[tid] = rv[tid + st]; ri[tid] = ri[tid + st];
        }
      }
      __syncthreads();
    }
    if (tid == 0) {
      int idx = ri[0]; float v = rv[0];
      vals[b * KSEL + j]       = v;
      perm[b * KSEL + j]       = b * LROI + idx;
      out_perm_f[b * KSEL + j] = (float)(b * LROI + idx);
      out_sig[b * KSEL + j]    = sigmoidf_(v);
      sv[idx] = -3.4e38f;
    }
    __syncthreads();
  }
}

__global__ void k_gather_scale(const float* __restrict__ T, const int* __restrict__ perm,
                               const float* __restrict__ vals, float* __restrict__ out,
                               int rows, int cols)
{
  int i = blockIdx.x * blockDim.x + threadIdx.x;
  if (i >= rows * cols) return;
  int r = i / cols, c = i % cols;
  out[i] = T[(size_t)perm[r] * cols + c] * vals[r];
}

__global__ void k_subpt(const float* __restrict__ ew, float* __restrict__ out)
{
  int i = blockIdx.x * blockDim.x + threadIdx.x;
  if (i >= NB * SUBW) return;
  int b = i / SUBW, s = i % SUBW;
  out[i] = ew[(size_t)b * SEGW + EPER + s];
}

// ======================================================================
//                               host
// ======================================================================
enum {
  P_node_emb_w, P_node_emb_b, P_pe_lin_w, P_pe_lin_b, P_pe_norm_g, P_pe_norm_b,
  P_gine_le_w, P_gine_le_b, P_gine_l1_w, P_gine_l1_b, P_gine_l2_w, P_gine_l2_b,
  P_gps_in_w, P_gps_in_b, P_gps_out_w, P_gps_out_b,
  P_bn1_g, P_bn1_b, P_bn2_g, P_bn2_b, P_bn3_g, P_bn3_b,
  P_mlp1_w, P_mlp1_b, P_mlp2_w, P_mlp2_b,
  P_node_emb1_w, P_node_emb1_b, P_pe_lin1_w, P_pe_lin1_b, P_pe_norm1_g, P_pe_norm1_b,
  P_node_emb2_w, P_node_emb2_b, P_pe_lin2_w, P_pe_lin2_b, P_pe_norm2_g, P_pe_norm2_b,
  P_gat1_w, P_gat1_as, P_gat1_ad, P_gat1_bias,
  P_gat2_w, P_gat2_as, P_gat2_ad, P_gat2_bias,
  P_mha1_in_w, P_mha1_in_b, P_mha1_out_w, P_mha1_out_b,
  P_mha2_in_w, P_mha2_in_b, P_mha2_out_w, P_mha2_out_b,
  P_pool1_w, P_pool2_w,
  P_lin11_w, P_lin11_b, P_lin12_w, P_lin12_b, P_lin13_w, P_lin13_b,
  P_lin14_w, P_lin14_b, P_lin15_w, P_lin15_b,
  P_lin21_w, P_lin21_b, P_lin23_w, P_lin23_b, P_lin24_w, P_lin24_b, P_lin25_w, P_lin25_b,
  P_COUNT
};

extern "C" void kernel_launch(void* const* d_in, const int* in_sizes, int n_in,
                              void* d_out, int out_size, void* d_ws, size_t ws_size,
                              hipStream_t stream)
{
  const float* x  = (const float*)d_in[0];
  const int*   ei = (const int*)d_in[1];
  const float* ew = (const float*)d_in[2];
  const float* prm[P_COUNT];
  for (int i = 0; i < P_COUNT; ++i) prm[i] = (const float*)d_in[4 + i];
  const int* src = ei;
  const int* dst = ei + NEDGE;
  float* out = (float*)d_out;

  // bump allocator over workspace
  char* wsp = (char*)d_ws;
  size_t used = 0;
  auto alloc = [&](size_t nfl) -> float* {
    float* p = (float*)(wsp + used);
    used += ((nfl * sizeof(float)) + 255) & ~((size_t)255);
    return p;
  };

  const int Nn = NNODES;
  float* f_tmp64 = alloc((size_t)Nn * 64);
  float* f_x0    = alloc((size_t)Nn * HCC);
  float* f_agg   = alloc((size_t)Nn * HCC);
  float* f_t3    = alloc((size_t)Nn * 528);
  float* f_h     = alloc((size_t)Nn * HCC);
  float* f_qkv   = alloc((size_t)Nn * 792);
  float* f_P     = alloc((size_t)NB * NHEAD * LROI * LROI);
  float* f_ao    = alloc((size_t)Nn * HCC);
  float* f_a     = alloc((size_t)Nn * HCC);
  float* f_out   = alloc((size_t)Nn * HCC);
  float* f_pre   = alloc((size_t)Nn * HCC);
  float* f_xt    = alloc((size_t)Nn * HC2C);
  float* f_es    = alloc(Nn);
  float* f_ed    = alloc(Nn);
  float* f_nmax  = alloc(Nn);
  float* f_nsum  = alloc(Nn);
  float* f_e     = alloc(NSEG);
  float* f_ex    = alloc(NSEG);
  float* f_gout  = alloc((size_t)Nn * HC2C);
  float* f_x11   = alloc((size_t)Nn * HC2C);
  float* f_x22   = alloc((size_t)Nn * HC2C);
  float* f_q     = alloc((size_t)Nn * HC2C);
  float* f_k     = alloc((size_t)Nn * HC2C);
  float* f_v     = alloc((size_t)Nn * HC2C);
  float* f_all1  = alloc((size_t)Nn * HC2C);
  float* f_t1    = alloc((size_t)Nn * HC2C);
  float* f_t2    = alloc((size_t)Nn * HC2C);
  float* f_score = alloc(Nn);
  float* f_vals  = alloc(NB * KSEL);
  int*   f_perm  = (int*)alloc(NB * KSEL);
  float* f_pool  = alloc((size_t)NB * KSEL * HC2C);
  float* f_x1l   = alloc(NB * HC2C);
  float* f_sub1  = alloc(NB * HC2C);
  float* f_cat   = alloc(NB * 2 * HC2C);
  float* f_c1    = alloc(NB * HC2C);
  float* f_subpt = alloc(NB * SUBW);
  float* f_scal  = alloc(8);
  __bf16* g_Abf  = (__bf16*)alloc(3u * 1024u * 1024u);   // 6M bf16 elems
  __bf16* g_Wbf  = (__bf16*)alloc(3u * 1024u * 1024u);   // 6M bf16 elems
  __bf16* g_Qpk  = (__bf16*)alloc(2400000);              // 128*384*96 = 4.72M bf16
  __bf16* g_Kpk  = (__bf16*)alloc(2400000);
  __bf16* g_Vt   = (__bf16*)alloc(2400000);              // 128*128*288 = 4.72M bf16
  __bf16* g_Ppk  = (__bf16*)alloc(7200000);              // 128*384*288 = 14.2M bf16
  (void)ws_size; (void)in_sizes; (void)n_in; (void)out_size;

  auto ew1 = [&](size_t n) { return dim3((unsigned)((n + 255) / 256)); };

  auto gemm_full = [&](const __bf16* A, int lda, size_t sA,
                       const __bf16* W, int ldw, size_t sW,
                       const float* bias, const float* res, int ldres,
                       float* C, int ldc, size_t sCb, size_t sCh, int nh,
                       int M, int Nc, int Kp, int act, float scale, int nz) {
    dim3 g((Nc + TN - 1) / TN, (M + TM - 1) / TM, nz);
    k_gemm_bf16<<<g, dim3(256), 0, stream>>>(A, lda, sA, W, ldw, sW, bias, res, ldres,
                                             C, ldc, sCb, sCh, nh, M, Nc, Kp, act, scale);
  };

  // pack fp32 operands to padded bf16, then run the async-WMMA GEMM
  auto gemm = [&](const float* A, int lda, const float* W, int ldw, const float* bias,
                  const float* res, int ldres, float* C, int ldc,
                  int M, int Nc, int Kd, int act) {
    int Mp = (M + 127) & ~127, Np = (Nc + 63) & ~63, Kp = (Kd + 31) & ~31;
    k_pack_bf16<<<ew1((size_t)Mp * Kp), 256, 0, stream>>>(A, lda, M, Kd, 0, g_Abf, Mp, Kp, 1);
    k_pack_bf16<<<ew1((size_t)Np * Kp), 256, 0, stream>>>(W, ldw, Nc, Kd, 0, g_Wbf, Np, Kp, 1);
    gemm_full(g_Abf, Kp, 0, g_Wbf, Kp, 0, bias, res, ldres, C, ldc, 0, 0, 1,
              M, Nc, Kp, act, 1.f, 1);
  };

  // batched WMMA attention: scores = scale*Q@K^T, softmax, O = P@V (per b,h)
  auto attention = [&](const float* Q, int ldq, int qc0,
                       const float* Km, int ldk, int kc0,
                       const float* V, int ldv, int vc0,
                       int dh, float scale, float* O, int ldo) {
    const int NZ  = NB * NHEAD;
    const int Lp  = 384;                    // 264 -> mult of 128 (A) & 64 (W)
    const int Dp  = (dh + 31) & ~31;        // K-pad: 96 / 64
    const int DpN = (dh + 63) & ~63;        // W-row pad for V^T: 128 / 64
    const int Kc  = 288;                    // 264 -> mult of 32
    k_pack_head<<<ew1((size_t)NZ * Lp * Dp), 256, 0, stream>>>(Q, ldq, qc0, dh, g_Qpk, Lp, Dp, NZ, 0);
    k_pack_head<<<ew1((size_t)NZ * Lp * Dp), 256, 0, stream>>>(Km, ldk, kc0, dh, g_Kpk, Lp, Dp, NZ, 0);
    gemm_full(g_Qpk, Dp, (size_t)Lp * Dp, g_Kpk, Dp, (size_t)Lp * Dp, nullptr, nullptr, 0,
              f_P, LROI, (size_t)LROI * LROI, 0, 1, LROI, LROI, Dp, 0, scale, NZ);
    k_softmax_row<<<NZ * LROI, 256, 0, stream>>>(f_P);
    k_pack_head<<<ew1((size_t)NZ * DpN * Kc), 256, 0, stream>>>(V, ldv, vc0, dh, g_Vt, DpN, Kc, NZ, 1);
    k_pack_bf16<<<ew1((size_t)NZ * Lp * Kc), 256, 0, stream>>>(f_P, LROI, LROI, LROI,
                                                               (size_t)LROI * LROI, g_Ppk, Lp, Kc, NZ);
    gemm_full(g_Ppk, Kc, (size_t)Lp * Kc, g_Vt, Kc, (size_t)DpN * Kc, nullptr, nullptr, 0,
              O, ldo, (size_t)LROI * ldo, (size_t)dh, NHEAD,
              LROI, dh, Kc, 0, 1.f, NZ);
  };

  // ---- stage 0: x0 = [x[:,:264]@We^T + be | sigmoid(bn(pro3)@Wpe^T+bpe)]
  k_bn_in<<<ew1(Nn * 64), 256, 0, stream>>>(x + 392, XW, prm[P_pe_norm_g], prm[P_pe_norm_b], f_tmp64, Nn, 64);
  gemm(x, XW, prm[P_node_emb_w], HCC, prm[P_node_emb_b], nullptr, 0, f_x0, HCC, Nn, 232, HCC, 0);
  gemm(f_tmp64, 64, prm[P_pe_lin_w], 64, prm[P_pe_lin_b], nullptr, 0, f_x0 + 232, HCC, Nn, 32, 64, 2);

  // ---- GINE conv
  hipMemsetAsync(f_agg, 0, (size_t)Nn * HCC * sizeof(float), stream);
  k_gine_msg<<<NEDGE, 256, 0, stream>>>(f_x0, src, dst, ew, prm[P_gine_le_w], prm[P_gine_le_b], f_agg);
  k_add<<<ew1(Nn * HCC), 256, 0, stream>>>(f_x0, f_agg, f_agg, Nn * HCC);
  gemm(f_agg, HCC, prm[P_gine_l1_w], HCC, prm[P_gine_l1_b], nullptr, 0, f_t3, HCC, Nn, HCC, HCC, 1);
  gemm(f_t3, HCC, prm[P_gine_l2_w], HCC, prm[P_gine_l2_b], nullptr, 0, f_h, HCC, Nn, HCC, HCC, 0);
  k_bn_res<<<ew1(Nn * HCC), 256, 0, stream>>>(f_h, f_x0, prm[P_bn1_g], prm[P_bn1_b], f_h, Nn, HCC, 0);

  // ---- GPS self-attention (dh = 66) on WMMA
  gemm(f_x0, HCC, prm[P_gps_in_w], HCC, prm[P_gps_in_b], nullptr, 0, f_qkv, 792, Nn, 792, HCC, 0);
  attention(f_qkv, 792, 0, f_qkv, 792, 264, f_qkv, 792, 528, 66, 1.0f / sqrtf(66.0f), f_ao, HCC);
  gemm(f_ao, HCC, prm[P_gps_out_w], HCC, prm[P_gps_out_b], nullptr, 0, f_a, HCC, Nn, HCC, HCC, 0);
  k_bn_res<<<ew1(Nn * HCC), 256, 0, stream>>>(f_a, f_x0, prm[P_bn2_g], prm[P_bn2_b], f_a, Nn, HCC, 0);
  k_add<<<ew1(Nn * HCC), 256, 0, stream>>>(f_h, f_a, f_out, Nn * HCC);
  gemm(f_out, HCC, prm[P_mlp1_w], HCC, prm[P_mlp1_b], nullptr, 0, f_t3, 528, Nn, 528, HCC, 1);
  gemm(f_t3, 528, prm[P_mlp2_w], 528, prm[P_mlp2_b], f_out, HCC, f_out, HCC, Nn, HCC, 528, 0);
  k_bn_res<<<ew1(Nn * HCC), 256, 0, stream>>>(f_out, nullptr, prm[P_bn3_g], prm[P_bn3_b], f_out, Nn, HCC, 1);

  // ---- GAT branches
  auto gat_branch = [&](int Wemb, int Bemb, int Wpe, int Bpe, int Gg, int Gb,
                        int Ww, int Was, int Wad, int Wbias, int xoff, float* xout) {
    gemm(f_out, HCC, prm[Wemb], HCC, prm[Bemb], nullptr, 0, f_pre, HCC, Nn, 232, HCC, 0);
    k_bn_in<<<ew1(Nn * 64), 256, 0, stream>>>(x + xoff, XW, prm[Gg], prm[Gb], f_tmp64, Nn, 64);
    gemm(f_tmp64, 64, prm[Wpe], 64, prm[Bpe], nullptr, 0, f_pre + 232, HCC, Nn, 32, 64, 2);
    gemm(f_pre, HCC, prm[Ww], HCC, nullptr, nullptr, 0, f_xt, HC2C, Nn, HC2C, HCC, 0);
    k_rowdot2<<<ew1(Nn), 256, 0, stream>>>(f_xt, prm[Was], prm[Wad], f_es, f_ed, Nn, HC2C);
    k_fill<<<ew1(Nn), 256, 0, stream>>>(f_nmax, -3.4e38f, Nn);
    k_gat_edge1<<<ew1(NSEG), 256, 0, stream>>>(src, dst, f_es, f_ed, f_e, f_nmax);
    hipMemsetAsync(f_nsum, 0, Nn * sizeof(float), stream);
    k_gat_edge2<<<ew1(NSEG), 256, 0, stream>>>(dst, f_e, f_nmax, f_ex, f_nsum);
    hipMemsetAsync(f_gout, 0, (size_t)Nn * HC2C * sizeof(float), stream);
    k_gat_scatter<<<NSEG, 256, 0, stream>>>(src, dst, f_ex, f_nsum, f_xt, f_gout);
    k_gat_fin<<<ew1(Nn * HC2C), 256, 0, stream>>>(f_gout, prm[Wbias], xout, Nn, HC2C);
  };
  gat_branch(P_node_emb1_w, P_node_emb1_b, P_pe_lin1_w, P_pe_lin1_b, P_pe_norm1_g, P_pe_norm1_b,
             P_gat1_w, P_gat1_as, P_gat1_ad, P_gat1_bias, 264, f_x11);
  gat_branch(P_node_emb2_w, P_node_emb2_b, P_pe_lin2_w, P_pe_lin2_b, P_pe_norm2_g, P_pe_norm2_b,
             P_gat2_w, P_gat2_as, P_gat2_ad, P_gat2_bias, 328, f_x22);

  // ---- MHA1 (q from x11, k/v from x22); mha2 is dead code in the reference.
  gemm(f_x11, HC2C, prm[P_mha1_in_w],              HC2C, prm[P_mha1_in_b],       nullptr, 0, f_q, HC2C, Nn, HC2C, HC2C, 0);
  gemm(f_x22, HC2C, prm[P_mha1_in_w] + 256 * HC2C, HC2C, prm[P_mha1_in_b] + 256, nullptr, 0, f_k, HC2C, Nn, HC2C, HC2C, 0);
  gemm(f_x22, HC2C, prm[P_mha1_in_w] + 512 * HC2C, HC2C, prm[P_mha1_in_b] + 512, nullptr, 0, f_v, HC2C, Nn, HC2C, HC2C, 0);
  attention(f_q, HC2C, 0, f_k, HC2C, 0, f_v, HC2C, 0, 64, 0.125f, f_ao, HC2C);
  gemm(f_ao, HC2C, prm[P_mha1_out_w], HC2C, prm[P_mha1_out_b], nullptr, 0, f_all1, HC2C, Nn, HC2C, HC2C, 0);
  k_attn_mean<<<ew1((size_t)NB * LROI * LROI), 256, 0, stream>>>(f_P, out + O_ATT);

  // ---- shared sub-feature path
  k_subpt<<<ew1(NB * SUBW), 256, 0, stream>>>(ew, f_subpt);
  gemm(f_subpt, SUBW, prm[P_lin12_w], SUBW, prm[P_lin12_b], nullptr, 0, f_sub1, HC2C, NB, HC2C, SUBW, 1);
  k_copy<<<ew1(NB * HC2C), 256, 0, stream>>>(f_sub1, out + O_SUB1, NB * HC2C);
  k_copy<<<ew1(NB * HC2C), 256, 0, stream>>>(f_sub1, out + O_SUB2, NB * HC2C);

  // ---- pooling branch helper
  auto pool_branch = [&](float* xin, int Wpool, int Wl1, int Bl1, int Wl3, int Bl3,
                         int Wl4, int Bl4, int dout4, float* tbuf,
                         int o_xl, int o_perm, int o_sc, int o_xo) {
    k_add<<<ew1(Nn * HC2C), 256, 0, stream>>>(f_all1, xin, tbuf, Nn * HC2C);
    k_vecnorm<<<1, 256, 0, stream>>>(prm[Wpool], f_scal, HC2C);
    k_pool_score<<<ew1(Nn), 256, 0, stream>>>(tbuf, prm[Wpool], f_scal, f_score, Nn, HC2C);
    k_topk<<<NB, 256, 0, stream>>>(f_score, f_vals, f_perm, out + o_perm, out + o_sc);
    k_gather_scale<<<ew1((size_t)NB * KSEL * HC2C), 256, 0, stream>>>(tbuf, f_perm, f_vals, f_pool, NB * KSEL, HC2C);
    gemm(f_pool, KSEL * HC2C, prm[Wl1], KSEL * HC2C, prm[Bl1], nullptr, 0, f_x1l, HC2C, NB, HC2C, KSEL * HC2C, 1);
    k_copy<<<ew1(NB * HC2C), 256, 0, stream>>>(f_x1l, out + o_xl, NB * HC2C);
    k_copy2d<<<ew1(NB * HC2C), 256, 0, stream>>>(f_x1l, HC2C, f_cat, 2 * HC2C, NB, HC2C);
    k_copy2d<<<ew1(NB * HC2C), 256, 0, stream>>>(f_sub1, HC2C, f_cat + HC2C, 2 * HC2C, NB, HC2C);
    gemm(f_cat, 2 * HC2C, prm[Wl3], 2 * HC2C, prm[Bl3], nullptr, 0, f_c1, HC2C, NB, HC2C, 2 * HC2C, 1);
    gemm(f_c1, HC2C, prm[Wl4], HC2C, prm[Bl4], nullptr, 0, out + o_xo, dout4, NB, dout4, HC2C, 0);
  };
  pool_branch(f_x11, P_pool1_w, P_lin11_w, P_lin11_b, P_lin13_w, P_lin13_b,
              P_lin14_w, P_lin14_b, 1, f_t1, O_X1L, O_PERM1, O_SC1, O_X1O);
  pool_branch(f_x22, P_pool2_w, P_lin21_w, P_lin21_b, P_lin23_w, P_lin23_b,
              P_lin24_w, P_lin24_b, 2, f_t2, O_X2L, O_PERM2, O_SC2, O_X2O);

  // sp1 = sub1 @ lin15^T + b ; sp2 = sub2 @ lin25^T + b
  gemm(f_sub1, HC2C, prm[P_lin15_w], HC2C, prm[P_lin15_b], nullptr, 0, out + O_SP1, 1, NB, 1, HC2C, 0);
  gemm(f_sub1, HC2C, prm[P_lin25_w], HC2C, prm[P_lin25_b], nullptr, 0, out + O_SP2, 2, NB, 2, HC2C, 0);
}
